// PerceiverAttention_55963423867389
// MI455X (gfx1250) — matrix operand order repaired and checked by reading, not compiled, not run on hardware
//
#include <hip/hip_runtime.h>
#include <cmath>

#define B_   8
#define T_   8192
#define L_   256
#define D_   512
#define H_   8
#define HD_  64
#define BH_  (B_*H_)

// hd^-0.5 * log2(e): folded into Q so the online softmax can use exp2 (v_exp_f32)
#define ATT_SCALE (0.125f * 1.44269504088896340736f)

typedef __attribute__((ext_vector_type(16))) __bf16        v16bf;
typedef __attribute__((ext_vector_type(8)))  __bf16        v8bf;
typedef __attribute__((ext_vector_type(8)))  float         v8f;
typedef __attribute__((ext_vector_type(4)))  unsigned int  u32x4;
typedef __attribute__((ext_vector_type(4)))  int           i32x4;
typedef __attribute__((ext_vector_type(8)))  int           i32x8;

#if defined(__has_builtin)
#  if __has_builtin(__builtin_amdgcn_tensor_load_to_lds) && \
      __has_builtin(__builtin_amdgcn_s_wait_tensorcnt)
#    define USE_TDM 1
#  endif
#  if __has_builtin(__builtin_amdgcn_exp2f)
#    define EXP2F(x) __builtin_amdgcn_exp2f(x)
#  endif
#endif
#ifndef USE_TDM
#  define USE_TDM 0
#endif
#ifndef EXP2F
#  define EXP2F(x) exp2f(x)
#endif

union V16u { v16bf v; v8bf h[2]; };

static __device__ __forceinline__ v8f wmma_bf16(v16bf a, v16bf b, v8f c) {
  // v_wmma_f32_16x16x32_bf16: (neg_a, A, neg_b, B, c_mod, C, reuse_a, reuse_b)
  return __builtin_amdgcn_wmma_f32_16x16x32_bf16(false, a, false, b, (short)0, c,
                                                 false, false);
}

// B fragment (32x16): lane holds one column, K {kb..kb+7} then {kb+16..kb+23} -> 2x16B loads
static __device__ __forceinline__ v16bf load_bfrag(const __bf16* p) {
  V16u u;
  u.h[0] = *(const v8bf*)(p);
  u.h[1] = *(const v8bf*)(p + 16);
  return u.v;
}

// A fragment (16x32) from bf16 row-major: K {kb..kb+7} then {kb+16..kb+23}
static __device__ __forceinline__ v16bf load_afrag_bf(const __bf16* p) {
  V16u u;
  u.h[0] = *(const v8bf*)(p);
  u.h[1] = *(const v8bf*)(p + 16);
  return u.v;
}

// A fragment from fp32 row-major with inline cvt to bf16 (v_cvt_pk_bf16_f32)
static __device__ __forceinline__ v16bf load_afrag_f32(const float* p) {
  float4 f0 = *(const float4*)(p);
  float4 f1 = *(const float4*)(p + 4);
  float4 f2 = *(const float4*)(p + 16);
  float4 f3 = *(const float4*)(p + 20);
  v16bf a;
  a[0]  = (__bf16)f0.x; a[1]  = (__bf16)f0.y; a[2]  = (__bf16)f0.z; a[3]  = (__bf16)f0.w;
  a[4]  = (__bf16)f1.x; a[5]  = (__bf16)f1.y; a[6]  = (__bf16)f1.z; a[7]  = (__bf16)f1.w;
  a[8]  = (__bf16)f2.x; a[9]  = (__bf16)f2.y; a[10] = (__bf16)f2.z; a[11] = (__bf16)f2.w;
  a[12] = (__bf16)f3.x; a[13] = (__bf16)f3.y; a[14] = (__bf16)f3.z; a[15] = (__bf16)f3.w;
  return a;
}

template <bool ABF>
static __device__ __forceinline__ v16bf loadA(const float* Af, const __bf16* Ab, size_t off) {
  if constexpr (ABF) return load_afrag_bf(Ab + off);
  else               return load_afrag_f32(Af + off);
}

// 32(M) x 64(N) wave tile, ping-pong pipelined (two fragment register sets alternate
// roles; no rotation moves, no WMMA->VALU WAR hazards on fragment registers).
// Requires K % 64 == 0.
template <bool ABF>
static __device__ __forceinline__ void gemm_32x64(const float* Af, const __bf16* Ab, int lda,
                                                  const __bf16* Wt, int ldw, int K, int lane,
                                                  v8f acc[2][4]) {
  const int col  = lane & 15;
  const int ksel = (lane < 16) ? 0 : 1;
  const size_t offA0 = (size_t)col * lda + 8 * ksel;
  const size_t offA1 = (size_t)(16 + col) * lda + 8 * ksel;
  const __bf16* pB = Wt + (size_t)col * ldw + 8 * ksel;

  // even set (k0), odd set (k0+32)
  v16bf a0 = loadA<ABF>(Af, Ab, offA0);
  v16bf a1 = loadA<ABF>(Af, Ab, offA1);
  v16bf b0 = load_bfrag(pB + (size_t)0  * ldw);
  v16bf b1 = load_bfrag(pB + (size_t)16 * ldw);
  v16bf b2 = load_bfrag(pB + (size_t)32 * ldw);
  v16bf b3 = load_bfrag(pB + (size_t)48 * ldw);

  for (int k0 = 0; k0 < K; k0 += 64) {
    // issue odd-set loads (k0+32), then consume even set
    const int km = k0 + 32;
    v16bf c0 = loadA<ABF>(Af, Ab, offA0 + km);
    v16bf c1 = loadA<ABF>(Af, Ab, offA1 + km);
    v16bf d0 = load_bfrag(pB + (size_t)0  * ldw + km);
    v16bf d1 = load_bfrag(pB + (size_t)16 * ldw + km);
    v16bf d2 = load_bfrag(pB + (size_t)32 * ldw + km);
    v16bf d3 = load_bfrag(pB + (size_t)48 * ldw + km);

    acc[0][0] = wmma_bf16(a0, b0, acc[0][0]);
    acc[1][0] = wmma_bf16(a1, b0, acc[1][0]);
    acc[0][1] = wmma_bf16(a0, b1, acc[0][1]);
    acc[1][1] = wmma_bf16(a1, b1, acc[1][1]);
    acc[0][2] = wmma_bf16(a0, b2, acc[0][2]);
    acc[1][2] = wmma_bf16(a1, b2, acc[1][2]);
    acc[0][3] = wmma_bf16(a0, b3, acc[0][3]);
    acc[1][3] = wmma_bf16(a1, b3, acc[1][3]);

    // issue even-set loads for next pair (k0+64), then consume odd set
    const int kn = k0 + 64;
    if (kn < K) {
      a0 = loadA<ABF>(Af, Ab, offA0 + kn);
      a1 = loadA<ABF>(Af, Ab, offA1 + kn);
      b0 = load_bfrag(pB + (size_t)0  * ldw + kn);
      b1 = load_bfrag(pB + (size_t)16 * ldw + kn);
      b2 = load_bfrag(pB + (size_t)32 * ldw + kn);
      b3 = load_bfrag(pB + (size_t)48 * ldw + kn);
    }
    acc[0][0] = wmma_bf16(c0, d0, acc[0][0]);
    acc[1][0] = wmma_bf16(c1, d0, acc[1][0]);
    acc[0][1] = wmma_bf16(c0, d1, acc[0][1]);
    acc[1][1] = wmma_bf16(c1, d1, acc[1][1]);
    acc[0][2] = wmma_bf16(c0, d2, acc[0][2]);
    acc[1][2] = wmma_bf16(c1, d2, acc[1][2]);
    acc[0][3] = wmma_bf16(c0, d3, acc[0][3]);
    acc[1][3] = wmma_bf16(c1, d3, acc[1][3]);
  }
}

#if USE_TDM
// Tensor Data Mover: 2D tile load (2-byte elements) per CDNA5 ISA.
// D# group0: count=1 | lds_addr | global_addr[56:0] | type=2
// D# group1: data_size=1(2B), tensor_dim0/1, tile_dim0/1, tensor_dim0_stride
static __device__ __forceinline__ void tdm_load_2d(unsigned lds_off, const void* gsrc,
                                                   unsigned dim0, unsigned dim1,
                                                   unsigned stride0,
                                                   unsigned tile0, unsigned tile1) {
  unsigned long long ga = (unsigned long long)(size_t)gsrc;
  u32x4 g0;
  g0[0] = 1u;                                                      // count=1
  g0[1] = lds_off;                                                 // LDS byte address
  g0[2] = (unsigned)ga;                                            // global_addr[31:0]
  g0[3] = (unsigned)((ga >> 32) & 0x01FFFFFFull) | 0x80000000u;    // addr[56:32]|type=2
  i32x8 g1;
  g1[0] = (int)(1u << 16);                                         // data_size = 2 bytes
  g1[1] = (int)((dim0 & 0xFFFFu) << 16);                           // tensor_dim0 lo16
  g1[2] = (int)(((dim0 >> 16) & 0xFFFFu) | ((dim1 & 0xFFFFu) << 16));
  g1[3] = (int)(((dim1 >> 16) & 0xFFFFu) | ((tile0 & 0xFFFFu) << 16));
  g1[4] = (int)(tile1 & 0xFFFFu);                                  // tile_dim1, tile_dim2=0
  g1[5] = (int)stride0;                                            // dim0_stride[31:0]
  g1[6] = 0;
  g1[7] = 0;
  i32x4 z4 = {0, 0, 0, 0};
#if __clang_major__ >= 23
  i32x8 z8 = {0, 0, 0, 0, 0, 0, 0, 0};
  __builtin_amdgcn_tensor_load_to_lds(g0, g1, z4, z4, z8, 0);
#else
  __builtin_amdgcn_tensor_load_to_lds(g0, g1, z4, z4, 0);
#endif
}
#endif

// ---------------- weight transpose + fp32 -> bf16 ----------------
__global__ __launch_bounds__(256) void transpose_convert(const float* __restrict__ W,
                                                         __bf16* __restrict__ Wt,
                                                         int in_dim, int out_dim) {
  int idx = blockIdx.x * blockDim.x + threadIdx.x;
  if (idx >= in_dim * out_dim) return;
  int o = idx / in_dim, i = idx - o * in_dim;
  Wt[idx] = (__bf16)W[(size_t)i * out_dim + o];  // Wt[o][i] = W[i][o]
}

// ---------------- q projection: Qh[bh][l][hd] = (query@Wq + bq) * ATT_SCALE ----
__global__ __launch_bounds__(128) void qproj_kernel(const float* __restrict__ query,
                                                    const __bf16* __restrict__ WqT,
                                                    const float* __restrict__ bq,
                                                    __bf16* __restrict__ Qh) {
  const int lane = threadIdx.x & 31;
  const int w    = threadIdx.x >> 5;
  const int tile = blockIdx.x * 4 + w;          // 512 tiles: (M/32)=64 x (N/64)=8
  const int nt = tile & 7, mt = tile >> 3;
  const int col = lane & 15, ksel = (lane < 16) ? 0 : 1;

  v8f acc[2][4] = {};
  gemm_32x64<false>(query + (size_t)mt * 32 * D_, nullptr, D_,
                    WqT + (size_t)nt * 64 * D_, D_, D_, lane, acc);

  const int h = nt;
#pragma unroll
  for (int sub = 0; sub < 2; ++sub)
#pragma unroll
    for (int j = 0; j < 4; ++j) {
      const int hd = j * 16 + col;
#pragma unroll
      for (int r = 0; r < 8; ++r) {
        int gm = mt * 32 + sub * 16 + r + ksel * 8;   // row in [0, B*L)
        int b = gm >> 8, l = gm & 255;
        float v = (acc[sub][j][r] + bq[h * 64 + hd]) * ATT_SCALE;
        Qh[(((size_t)(b * H_ + h)) * L_ + l) * HD_ + hd] = (__bf16)v;
      }
    }
}

// ---------------- kv projection: K[bh][t][d], Vt[bh][d][t] (bf16) ----------------
// Epilogue bounces the 32x64 result tile through a per-wave LDS staging buffer so the
// 134MB K/V write stream goes out as coalesced b128 stores instead of 2B scatter.
__global__ __launch_bounds__(128) void kvproj_kernel(const float* __restrict__ x,
                                                     const __bf16* __restrict__ WkvT,
                                                     const float* __restrict__ bkv,
                                                     __bf16* __restrict__ Kb,
                                                     __bf16* __restrict__ Vt) {
  __shared__ __align__(128) __bf16 stg[4][32 * 64];   // 16KB, one 4KB tile per wave
  const int lane = threadIdx.x & 31;
  const int w    = threadIdx.x >> 5;
  const int tile = blockIdx.x * 4 + w;          // 32768 tiles: (M/32)=2048 x (N/64)=16
  const int nt = tile & 15, mt = tile >> 4;
  const int col = lane & 15, ksel = (lane < 16) ? 0 : 1;
  __bf16* my = stg[w];

  v8f acc[2][4] = {};
  gemm_32x64<false>(x + (size_t)mt * 32 * D_, nullptr, D_,
                    WkvT + (size_t)nt * 64 * D_, D_, D_, lane, acc);

  const bool isV = (nt >= 8);
  const int h = nt & 7;
  const int gm0 = mt * 32;                       // first row of tile, in [0, B*T)
  const int b  = gm0 >> 13;                      // tiles never straddle a batch (8192%32==0)
  const int t0 = gm0 & (T_ - 1);

  if (!isV) {
    // stage [t_local 32][d 64]; 32 consecutive t rows * 64 d are contiguous in Kb
#pragma unroll
    for (int sub = 0; sub < 2; ++sub)
#pragma unroll
      for (int j = 0; j < 4; ++j)
#pragma unroll
        for (int r = 0; r < 8; ++r)
          my[(sub * 16 + r + ksel * 8) * 64 + j * 16 + col] =
              (__bf16)(acc[sub][j][r] + bkv[nt * 64 + j * 16 + col]);
    v8bf* dst = (v8bf*)(Kb + (((size_t)(b * H_ + h)) * T_ + t0) * HD_);
    const v8bf* src = (const v8bf*)my;
#pragma unroll
    for (int i = 0; i < 8; ++i) dst[lane + 32 * i] = src[lane + 32 * i];  // 4KB flat copy
  } else {
    // stage transposed [d 64][t_local 32]; each d-row is 64B contiguous in Vt
#pragma unroll
    for (int sub = 0; sub < 2; ++sub)
#pragma unroll
      for (int j = 0; j < 4; ++j)
#pragma unroll
        for (int r = 0; r < 8; ++r)
          my[(j * 16 + col) * 32 + sub * 16 + r + ksel * 8] =
              (__bf16)(acc[sub][j][r] + bkv[nt * 64 + j * 16 + col]);
    __bf16* vbase = Vt + ((size_t)(b * H_ + h)) * HD_ * T_ + t0;
#pragma unroll
    for (int i = 0; i < 8; ++i) {
      int c = lane + 32 * i;                     // 256 chunks of 16B
      int d = c >> 2, part = c & 3;
      *(v8bf*)(vbase + (size_t)d * T_ + part * 8) = *(const v8bf*)(my + d * 32 + part * 8);
    }
  }
}

// ---------------- flash attention with TDM-staged K/V tiles in LDS ----------------
// smem map (single static LDS block; its base is the kernel's only LDS alloc -> off 0):
//   [ K tile buf0 8KB | V tile buf0 8KB | K buf1 8KB | V buf1 8KB | P staging 8KB ]
#define SM_K(b) ((unsigned)(b) * 16384u)
#define SM_V(b) ((unsigned)(b) * 16384u + 8192u)
#define SM_P    32768u
#define SM_TOT  40960u

__global__ __launch_bounds__(128) void attn_kernel(const __bf16* __restrict__ Qh,
                                                   const __bf16* __restrict__ Kb,
                                                   const __bf16* __restrict__ Vt,
                                                   __bf16* __restrict__ Obf) {
  __shared__ __align__(1024) char smem[SM_TOT];
  const int tid  = threadIdx.x;
  const int lane = tid & 31;
  const int wid  = tid >> 5;
  const int bh   = blockIdx.x;                   // 0..63
  const int b = bh >> 3, h = bh & 7;
  const int col = lane & 15, ksel = (lane < 16) ? 0 : 1;
  __bf16* Pw = (__bf16*)(smem + SM_P) + wid * 16 * 64;

  const int qbase = blockIdx.y * 64 + wid * 16;  // 16 query rows per wave
  const __bf16* Qb = Qh + ((size_t)bh * L_ + qbase) * HD_;
  const __bf16* Kh = Kb + (size_t)bh * T_ * HD_;
  const __bf16* Vh = Vt + (size_t)bh * HD_ * T_;

  const v16bf qa0 = load_afrag_bf(Qb + (size_t)col * HD_ + 0  + 8 * ksel);
  const v16bf qa1 = load_afrag_bf(Qb + (size_t)col * HD_ + 32 + 8 * ksel);

  float mrow[8], lrow[8];
  v8f o[4] = {};
#pragma unroll
  for (int r = 0; r < 8; ++r) { mrow[r] = -INFINITY; lrow[r] = 0.f; }

#if USE_TDM
  if (wid == 0) {   // prime buffer 0 (wave-level op; EXEC ignored by TDM)
    tdm_load_2d(SM_K(0), Kh, HD_, T_, HD_, HD_, 64);
    tdm_load_2d(SM_V(0), Vh, T_, HD_, T_, 64, HD_);
  }
#endif

  int buf = 0;
  for (int t0 = 0; t0 < T_; t0 += 64, buf ^= 1) {
#if USE_TDM
    if (wid == 0) {
      if (t0 + 64 < T_) {     // issue next tile into the other buffer, then wait
        tdm_load_2d(SM_K(buf ^ 1), Kh + (size_t)(t0 + 64) * HD_, HD_, T_, HD_, HD_, 64);
        tdm_load_2d(SM_V(buf ^ 1), Vh + (t0 + 64), T_, HD_, T_, 64, HD_);
        __builtin_amdgcn_s_wait_tensorcnt(2);  // TDM is in-order: current buf done
      } else {
        __builtin_amdgcn_s_wait_tensorcnt(0);
      }
    }
    __syncthreads();          // current buffer visible to all waves
    const unsigned kOff = SM_K(buf), vOff = SM_V(buf);
#else
    __syncthreads();          // prior readers done before overwrite
    {                         // cooperative synchronous staging (single buffer)
      const v8bf* sK = (const v8bf*)(Kh + (size_t)t0 * HD_);   // tile is contiguous
      v8bf*       dK = (v8bf*)(smem + SM_K(0));
#pragma unroll
      for (int i = 0; i < 4; ++i) dK[tid * 4 + i] = sK[tid * 4 + i];
      const int vr = tid >> 1, vh = (tid & 1) * 32;
      const v8bf* sV = (const v8bf*)(Vh + (size_t)vr * T_ + t0 + vh);
      v8bf*       dV = (v8bf*)(smem + SM_V(0) + ((size_t)vr * 64 + vh) * 2);
#pragma unroll
      for (int i = 0; i < 4; ++i) dV[i] = sV[i];
    }
    __syncthreads();
    const unsigned kOff = SM_K(0), vOff = SM_V(0);
#endif
    const __bf16* Klds = (const __bf16*)(smem + kOff);   // [t 64][d 64]
    const __bf16* Vlds = (const __bf16*)(smem + vOff);   // [d 64][t 64]

    // ---- scores: S(16x64) = Q . K^T, B-frags from LDS ----
    v8f s[4];
#pragma unroll
    for (int j = 0; j < 4; ++j) {
      const __bf16* kp = Klds + (size_t)(j * 16 + col) * 64 + 8 * ksel;
      v8f z = {};
      z    = wmma_bf16(qa0, load_bfrag(kp),      z);
      s[j] = wmma_bf16(qa1, load_bfrag(kp + 32), z);
    }

    // preload all V B-frags now so ds_load latency hides under the softmax VALU work
    v16bf vb[4][2];
#pragma unroll
    for (int j = 0; j < 4; ++j) {
      const __bf16* vp = Vlds + (size_t)(j * 16 + col) * 64 + 8 * ksel;
      vb[j][0] = load_bfrag(vp);
      vb[j][1] = load_bfrag(vp + 32);
    }

    // ---- online softmax (row reductions across 16-lane halves) ----
    float mx[8];
#pragma unroll
    for (int r = 0; r < 8; ++r)
      mx[r] = fmaxf(fmaxf(s[0][r], s[1][r]), fmaxf(s[2][r], s[3][r]));
#pragma unroll
    for (int r = 0; r < 8; ++r) {
#pragma unroll
      for (int d = 1; d < 16; d <<= 1) mx[r] = fmaxf(mx[r], __shfl_xor(mx[r], d, 32));
    }
    float alpha[8];
#pragma unroll
    for (int r = 0; r < 8; ++r) {
      float mn = fmaxf(mrow[r], mx[r]);
      alpha[r] = EXP2F(mrow[r] - mn);
      mrow[r]  = mn;
    }
#pragma unroll
    for (int j = 0; j < 4; ++j)
#pragma unroll
      for (int r = 0; r < 8; ++r) s[j][r] = EXP2F(s[j][r] - mrow[r]);
#pragma unroll
    for (int r = 0; r < 8; ++r) {
      float t = s[0][r] + s[1][r] + s[2][r] + s[3][r];
#pragma unroll
      for (int d = 1; d < 16; d <<= 1) t += __shfl_xor(t, d, 32);
      lrow[r] = lrow[r] * alpha[r] + t;
    }
#pragma unroll
    for (int j = 0; j < 4; ++j)
#pragma unroll
      for (int r = 0; r < 8; ++r) o[j][r] *= alpha[r];

    // ---- C-frag -> A-frag relayout of P through per-wave LDS region ----
    const int mb = ksel * 8;
#pragma unroll
    for (int j = 0; j < 4; ++j)
#pragma unroll
      for (int r = 0; r < 8; ++r)
        Pw[(r + mb) * 64 + j * 16 + col] = (__bf16)s[j][r];

    const v16bf pa0 = load_afrag_bf(Pw + col * 64 + 0  + 8 * ksel);
    const v16bf pa1 = load_afrag_bf(Pw + col * 64 + 32 + 8 * ksel);

    // ---- O(16x64) += P(16x64) . V(64x64), V-tile transposed [d][t] in LDS ----
#pragma unroll
    for (int j = 0; j < 4; ++j) {
      o[j] = wmma_bf16(pa0, vb[j][0], o[j]);
      o[j] = wmma_bf16(pa1, vb[j][1], o[j]);
    }
#if USE_TDM
    __syncthreads();          // all waves done with this buffer before next overwrite
#endif
  }

  // ---- epilogue: normalize, stage in Pw, write 128B-coalesced rows of Obf ----
#pragma unroll
  for (int r = 0; r < 8; ++r) {
    float inv = (lrow[r] > 0.f) ? 1.0f / lrow[r] : 0.f;
#pragma unroll
    for (int j = 0; j < 4; ++j)
      Pw[(r + ksel * 8) * 64 + j * 16 + col] = (__bf16)(o[j][r] * inv);
  }
  {
    __bf16* ob = Obf + ((size_t)b * L_ + qbase) * D_ + h * 64;
#pragma unroll
    for (int i = 0; i < 4; ++i) {
      int c = lane + 32 * i;                     // 128 chunks of 16B (16 rows x 128B)
      int m = c >> 3, part = c & 7;
      *(v8bf*)(ob + (size_t)m * D_ + part * 8) = *(const v8bf*)(Pw + m * 64 + part * 8);
    }
  }
}

// ---------------- output projection: out = Obf @ Wproj + bproj (fp32 out) ------
__global__ __launch_bounds__(128) void proj_kernel(const __bf16* __restrict__ Obf,
                                                   const __bf16* __restrict__ WprojT,
                                                   const float* __restrict__ bproj,
                                                   float* __restrict__ out) {
  const int lane = threadIdx.x & 31;
  const int w    = threadIdx.x >> 5;
  const int tile = blockIdx.x * 4 + w;          // 512 tiles: 64 x 8
  const int nt = tile & 7, mt = tile >> 3;
  const int col = lane & 15, ksel = (lane < 16) ? 0 : 1;

  v8f acc[2][4] = {};
  gemm_32x64<true>(nullptr, Obf + (size_t)mt * 32 * D_, D_,
                   WprojT + (size_t)nt * 64 * D_, D_, D_, lane, acc);

#pragma unroll
  for (int sub = 0; sub < 2; ++sub)
#pragma unroll
    for (int j = 0; j < 4; ++j) {
      const int n = nt * 64 + j * 16 + col;
#pragma unroll
      for (int r = 0; r < 8; ++r) {
        int gm = mt * 32 + sub * 16 + r + ksel * 8;
        out[(size_t)gm * D_ + n] = acc[sub][j][r] + bproj[n];
      }
    }
}

extern "C" void kernel_launch(void* const* d_in, const int* in_sizes, int n_in,
                              void* d_out, int out_size, void* d_ws, size_t ws_size,
                              hipStream_t stream) {
  const float* x     = (const float*)d_in[0];
  const float* query = (const float*)d_in[1];
  const float* Wq    = (const float*)d_in[2];
  const float* bq    = (const float*)d_in[3];
  const float* Wkv   = (const float*)d_in[4];
  const float* bkv   = (const float*)d_in[5];
  const float* Wproj = (const float*)d_in[6];
  const float* bproj = (const float*)d_in[7];
  float* out = (float*)d_out;

  char* ws = (char*)d_ws;
  size_t off = 0;
  auto take = [&](size_t bytes) { size_t o = off; off = (off + bytes + 255) & ~(size_t)255; return o; };
  __bf16* WqT    = (__bf16*)(ws + take((size_t)D_ * D_ * 2));
  __bf16* WkvT   = (__bf16*)(ws + take((size_t)2 * D_ * D_ * 2));
  __bf16* WprojT = (__bf16*)(ws + take((size_t)D_ * D_ * 2));
  __bf16* Qh     = (__bf16*)(ws + take((size_t)BH_ * L_ * HD_ * 2));
  __bf16* Kb     = (__bf16*)(ws + take((size_t)BH_ * T_ * HD_ * 2));
  __bf16* Vt     = (__bf16*)(ws + take((size_t)BH_ * HD_ * T_ * 2));
  __bf16* Obf    = (__bf16*)(ws + take((size_t)B_ * L_ * D_ * 2));

  // 1) weights -> [out][in] bf16
  transpose_convert<<<(D_ * D_ + 255) / 256, 256, 0, stream>>>(Wq, WqT, D_, D_);
  transpose_convert<<<(2 * D_ * D_ + 255) / 256, 256, 0, stream>>>(Wkv, WkvT, D_, 2 * D_);
  transpose_convert<<<(D_ * D_ + 255) / 256, 256, 0, stream>>>(Wproj, WprojT, D_, D_);

  // 2) q projection (softmax scale folded in)
  qproj_kernel<<<(B_ * L_ / 32) * (D_ / 64) / 4, 128, 0, stream>>>(query, WqT, bq, Qh);

  // 3) kv projection -> K [bh][t][d], V transposed [bh][d][t]
  kvproj_kernel<<<(B_ * T_ / 32) * (2 * D_ / 64) / 4, 128, 0, stream>>>(x, WkvT, bkv, Kb, Vt);

  // 4) flash attention: one (b,h) per blockIdx.x, 64 queries per block (16/wave)
  attn_kernel<<<dim3(BH_, L_ / 64), 128, 0, stream>>>(Qh, Kb, Vt, Obf);

  // 5) out projection (fp32 result)
  proj_kernel<<<(B_ * L_ / 32) * (D_ / 64) / 4, 128, 0, stream>>>(Obf, WprojT, bproj, out);
}
